// Model_24644522344786
// MI455X (gfx1250) — matrix operand order repaired and checked by reading, not compiled, not run on hardware
//
#include <hip/hip_runtime.h>
#include <hip/hip_bf16.h>

typedef __attribute__((ext_vector_type(16))) _Float16 v16h;
typedef __attribute__((ext_vector_type(8)))  _Float16 v8h;
typedef __attribute__((ext_vector_type(8)))  float    v8f;

#define N_GENES 20000
#define N_TFS   1500
#define N_TFS_P 1504          // h0 K padded to multiple of 32
#define N_CONN  200000
#define N_PERT  128
#define N_HID   64
#define BATCH   1024

#define ROWS       8          // batch rows per scatter block
#define LDS_STRIDE 1512       // 1512 mod 64 = 40; gcd(40,64)=8 -> 8 rows hit 8 banks

// ---------------------------------------------------------------------------
// WMMA helper: D = A(16x32 f16) * B(32x16 f16) + C(16x16 f32)
// ---------------------------------------------------------------------------
__device__ __forceinline__ v8f wmma_f16(v16h a, v16h b, v8f c) {
    return __builtin_amdgcn_wmma_f32_16x16x32_f16(
        /*neg_a=*/false, a, /*neg_b=*/false, b,
        /*c_mod=*/(short)0, c, /*reuse_a=*/false, /*reuse_b=*/false);
}

// A-operand fragment load (16-bit A 16x32 layout, wave32):
// lane L: M = L&15; elements j=0..7  -> K = k0 + (L>>4)*8 + j
//         elements j=8..15 -> K = k0 + 16 + (L>>4)*8 + (j-8)
__device__ __forceinline__ v16h load_a_frag(const _Float16* base, int ldk,
                                            int m0, int k0, int lane) {
    int m    = m0 + (lane & 15);
    int koff = (lane >> 4) * 8;
    const _Float16* p = base + (size_t)m * ldk + k0 + koff;
    v8h lo = *(const v8h*)(p);
    v8h hi = *(const v8h*)(p + 16);
    v16h a;
#pragma unroll
    for (int i = 0; i < 8; ++i) { a[i] = lo[i]; a[i + 8] = hi[i]; }
    return a;
}

// ---------------------------------------------------------------------------
// Kernel 0: pack f32 weight matrix (K x N, row-major) into f16 B-operand
// layout. out element index: (((nt*KT + kt)*32 + L)*16 + j)
//   K_local = (j>>3)*16 + (L>>4)*8 + (j&7) ; N = nt*16 + (L&15) ; K >= Kreal zero-padded.
// ---------------------------------------------------------------------------
__global__ void pack_b_kernel(const float* __restrict__ W, _Float16* __restrict__ out,
                              int Kreal, int N, int KT) {
    size_t total = (size_t)(N / 16) * KT * 512;
    size_t tid = (size_t)blockIdx.x * 256 + threadIdx.x;
    if (tid >= total) return;
    int j = (int)(tid & 15);
    int L = (int)((tid >> 4) & 31);
    size_t rem = tid >> 9;
    int kt = (int)(rem % KT);
    int nt = (int)(rem / KT);
    int k = kt * 32 + (j >> 3) * 16 + (L >> 4) * 8 + (j & 7);
    int n = nt * 16 + (L & 15);
    out[tid] = (k < Kreal) ? (_Float16)W[(size_t)k * N + n] : (_Float16)0.0f;
}

// ---------------------------------------------------------------------------
// Kernel 1: sparse gather/scatter + PReLU(a0) -> h0 (f16, BATCH x N_TFS_P)
// 8 batch rows per block; f32 LDS accumulation (ds_add_f32).
//  - edge stream loaded as int4/float4 (4 edges per b128 triple)
//  - LDS row stride 1512 spreads the 8 row-copies over 8 distinct banks
// ---------------------------------------------------------------------------
__global__ void __launch_bounds__(256)
scatter_kernel(const float* __restrict__ x, const int* __restrict__ gene_idx,
               const int* __restrict__ tf_idx, const float* __restrict__ edge_w,
               const float* __restrict__ a0, _Float16* __restrict__ h0) {
    __shared__ float acc[ROWS * LDS_STRIDE];
    int tid = threadIdx.x;
    for (int i = tid; i < ROWS * LDS_STRIDE; i += 256) acc[i] = 0.0f;
    __syncthreads();

    int row0 = blockIdx.x * ROWS;
    const float* xbase = x + (size_t)row0 * N_GENES;

    const int4*   g4p = (const int4*)gene_idx;
    const int4*   t4p = (const int4*)tf_idx;
    const float4* w4p = (const float4*)edge_w;
    const int NV = N_CONN / 4;                 // 50000

    for (int v = tid; v < NV; v += 256) {
        int4   g4 = g4p[v];
        int4   t4 = t4p[v];
        float4 w4 = w4p[v];
#pragma unroll
        for (int s = 0; s < 4; ++s) {
            int   g = (s == 0) ? g4.x : (s == 1) ? g4.y : (s == 2) ? g4.z : g4.w;
            int   t = (s == 0) ? t4.x : (s == 1) ? t4.y : (s == 2) ? t4.z : t4.w;
            float w = (s == 0) ? w4.x : (s == 1) ? w4.y : (s == 2) ? w4.z : w4.w;
#pragma unroll
            for (int r = 0; r < ROWS; ++r) {
                float xv = xbase[(size_t)r * N_GENES + g];
                atomicAdd(&acc[r * LDS_STRIDE + t], xv * w);
            }
        }
    }
    __syncthreads();

    float slope = a0[0];
    for (int i = tid; i < ROWS * N_TFS_P; i += 256) {
        int r = i / N_TFS_P;
        int k = i - r * N_TFS_P;
        float v = acc[r * LDS_STRIDE + k];   // k in [1500,1504) stays 0
        v = (v >= 0.0f) ? v : slope * v;
        h0[(size_t)(row0 + r) * N_TFS_P + k] = (_Float16)v;
    }
}

// ---------------------------------------------------------------------------
// Kernel 2: h1 = f16( prelu(h0 @ W1 + b1, a1) ), K = 1504 (47 steps)
// block = 4 waves; wave w handles n-tile w; blockIdx.x = m-tile (64 total)
// Two independent accumulators (even/odd k-steps) to break WMMA chains.
// ---------------------------------------------------------------------------
__global__ void __launch_bounds__(128)
gemm1_kernel(const _Float16* __restrict__ h0, const _Float16* __restrict__ W1p,
             const float* __restrict__ b1, const float* __restrict__ a1,
             _Float16* __restrict__ h1) {
    int lane = threadIdx.x & 31;
    int nt   = threadIdx.x >> 5;         // 0..3
    int mt   = blockIdx.x;               // 0..63
    const int KT = N_TFS_P / 32;         // 47
    const _Float16* Wbase = W1p + (size_t)nt * KT * 512;

    v8f c0 = {}, c1 = {};
    int kt = 0;
    for (; kt + 1 < KT; kt += 2) {
        v16h a0 = load_a_frag(h0, N_TFS_P, mt * 16, kt * 32, lane);
        v16h b0 = *(const v16h*)(Wbase + ((size_t)kt * 32 + lane) * 16);
        v16h a1v = load_a_frag(h0, N_TFS_P, mt * 16, (kt + 1) * 32, lane);
        v16h b1v = *(const v16h*)(Wbase + ((size_t)(kt + 1) * 32 + lane) * 16);
        c0 = wmma_f16(a0, b0, c0);
        c1 = wmma_f16(a1v, b1v, c1);
    }
    { // tail (KT odd)
        v16h a0 = load_a_frag(h0, N_TFS_P, mt * 16, kt * 32, lane);
        v16h b0 = *(const v16h*)(Wbase + ((size_t)kt * 32 + lane) * 16);
        c0 = wmma_f16(a0, b0, c0);
    }

    float slope = a1[0];
    int   n     = nt * 16 + (lane & 15);
    float bias  = b1[n];
    int   half  = lane >> 4;
#pragma unroll
    for (int v = 0; v < 8; ++v) {
        int   m = mt * 16 + v + half * 8;
        float y = c0[v] + c1[v] + bias;
        y = (y >= 0.0f) ? y : slope * y;
        h1[(size_t)m * N_HID + n] = (_Float16)y;
    }
}

// ---------------------------------------------------------------------------
// Kernel 3: per 16-row stripe (4 waves, one n-tile per wave per stage):
//   h2 = prelu(h1 @ W2 + b2, a2) + emb[pert];  h3 = prelu(h2 @ Wd1 + bd1, ad)
// h2 staged f16 in LDS to re-load in A-fragment layout.
// ---------------------------------------------------------------------------
__global__ void __launch_bounds__(128)
mid_kernel(const _Float16* __restrict__ h1, const _Float16* __restrict__ W2p,
           const float* __restrict__ b2, const float* __restrict__ a2,
           const float* __restrict__ emb, const int* __restrict__ pert,
           const _Float16* __restrict__ Wd1p, const float* __restrict__ bd1,
           const float* __restrict__ ad, _Float16* __restrict__ h3) {
    __shared__ __align__(16) _Float16 lds[16 * N_HID];
    int lane = threadIdx.x & 31;
    int nt   = threadIdx.x >> 5;         // 0..3 : this wave's n-tile
    int mt   = blockIdx.x;               // 0..63
    int half = lane >> 4;
    int nl   = lane & 15;
    const int KT = N_HID / 32;           // 2
    float sa2 = a2[0], sad = ad[0];

    // stage 1: h2 tile nt -> LDS
    {
        v8f c0 = {}, c1 = {};
        v16h a0 = load_a_frag(h1, N_HID, mt * 16, 0, lane);
        v16h b0 = *(const v16h*)(W2p + ((size_t)(nt * KT + 0) * 32 + lane) * 16);
        v16h a1v = load_a_frag(h1, N_HID, mt * 16, 32, lane);
        v16h b1v = *(const v16h*)(W2p + ((size_t)(nt * KT + 1) * 32 + lane) * 16);
        c0 = wmma_f16(a0, b0, c0);
        c1 = wmma_f16(a1v, b1v, c1);

        int   n    = nt * 16 + nl;
        float bias = b2[n];
#pragma unroll
        for (int v = 0; v < 8; ++v) {
            int   mrow = v + half * 8;
            int   m    = mt * 16 + mrow;
            float y = c0[v] + c1[v] + bias;
            y = (y >= 0.0f) ? y : sa2 * y;
            y += emb[(size_t)pert[m] * N_HID + n];
            lds[mrow * N_HID + n] = (_Float16)y;
        }
    }
    __syncthreads();

    // stage 2: h3 tile nt = prelu(h2 @ Wd1 + bd1)
    {
        v8f c0 = {}, c1 = {};
        v16h a0 = load_a_frag(lds, N_HID, 0, 0, lane);
        v16h b0 = *(const v16h*)(Wd1p + ((size_t)(nt * KT + 0) * 32 + lane) * 16);
        v16h a1v = load_a_frag(lds, N_HID, 0, 32, lane);
        v16h b1v = *(const v16h*)(Wd1p + ((size_t)(nt * KT + 1) * 32 + lane) * 16);
        c0 = wmma_f16(a0, b0, c0);
        c1 = wmma_f16(a1v, b1v, c1);

        int   n    = nt * 16 + nl;
        float bias = bd1[n];
#pragma unroll
        for (int v = 0; v < 8; ++v) {
            int   m = mt * 16 + v + half * 8;
            float y = c0[v] + c1[v] + bias;
            y = (y >= 0.0f) ? y : sad * y;
            h3[(size_t)m * N_HID + n] = (_Float16)y;
        }
    }
}

// ---------------------------------------------------------------------------
// Kernel 4: out = h3 @ Wd2 + bd2  (1024 x 20000, f32). The bandwidth kernel.
// block = 8 waves; wave handles one n-tile; grid (64, ceil(1250/8)=157).
// Non-temporal stores: output is write-once, keep L2 for reused x / weights.
// ---------------------------------------------------------------------------
__global__ void __launch_bounds__(256)
gemm2_kernel(const _Float16* __restrict__ h3, const _Float16* __restrict__ Wd2p,
             const float* __restrict__ bd2, float* __restrict__ out) {
    int lane = threadIdx.x & 31;
    int w    = threadIdx.x >> 5;         // 0..7
    int mt   = blockIdx.x;               // 0..63
    int nt   = blockIdx.y * 8 + w;       // 0..1249
    if (nt >= N_GENES / 16) return;
    const int KT = N_HID / 32;           // 2

    v16h a0 = load_a_frag(h3, N_HID, mt * 16, 0, lane);
    v16h a1 = load_a_frag(h3, N_HID, mt * 16, 32, lane);
    v16h b0 = *(const v16h*)(Wd2p + ((size_t)(nt * KT + 0) * 32 + lane) * 16);
    v16h b1 = *(const v16h*)(Wd2p + ((size_t)(nt * KT + 1) * 32 + lane) * 16);
    v8f c0 = {}, c1 = {};
    c0 = wmma_f16(a0, b0, c0);          // independent WMMAs, dual-issue
    c1 = wmma_f16(a1, b1, c1);

    int   n    = nt * 16 + (lane & 15);
    float bias = bd2[n];
    int   half = lane >> 4;
#pragma unroll
    for (int v = 0; v < 8; ++v) {
        int m = mt * 16 + v + half * 8;
        __builtin_nontemporal_store(c0[v] + c1[v] + bias,
                                    &out[(size_t)m * N_GENES + n]);
    }
}

// ---------------------------------------------------------------------------
// Launch
// ---------------------------------------------------------------------------
static inline size_t align256(size_t v) { return (v + 255) & ~(size_t)255; }

extern "C" void kernel_launch(void* const* d_in, const int* in_sizes, int n_in,
                              void* d_out, int out_size, void* d_ws, size_t ws_size,
                              hipStream_t stream) {
    (void)in_sizes; (void)n_in; (void)out_size; (void)ws_size;
    const float* x        = (const float*)d_in[0];
    const int*   pert     = (const int*)  d_in[1];
    const int*   gene_idx = (const int*)  d_in[2];
    const int*   tf_idx   = (const int*)  d_in[3];
    const float* edge_w   = (const float*)d_in[4];
    const float* emb      = (const float*)d_in[5];
    const float* a0       = (const float*)d_in[6];
    const float* W1       = (const float*)d_in[7];
    const float* b1       = (const float*)d_in[8];
    const float* a1       = (const float*)d_in[9];
    const float* W2       = (const float*)d_in[10];
    const float* b2       = (const float*)d_in[11];
    const float* a2       = (const float*)d_in[12];
    const float* Wd1      = (const float*)d_in[13];
    const float* bd1      = (const float*)d_in[14];
    const float* ad       = (const float*)d_in[15];
    const float* Wd2      = (const float*)d_in[16];
    const float* bd2      = (const float*)d_in[17];
    float* out = (float*)d_out;

    // workspace layout
    char* ws = (char*)d_ws;
    size_t off = 0;
    _Float16* h0   = (_Float16*)(ws + off); off = align256(off + (size_t)BATCH * N_TFS_P * 2);
    _Float16* W1p  = (_Float16*)(ws + off); off = align256(off + (size_t)N_TFS_P * N_HID * 2);
    _Float16* h1   = (_Float16*)(ws + off); off = align256(off + (size_t)BATCH * N_HID * 2);
    _Float16* W2p  = (_Float16*)(ws + off); off = align256(off + (size_t)N_HID * N_HID * 2);
    _Float16* Wd1p = (_Float16*)(ws + off); off = align256(off + (size_t)N_HID * N_HID * 2);
    _Float16* h3   = (_Float16*)(ws + off); off = align256(off + (size_t)BATCH * N_HID * 2);
    _Float16* Wd2p = (_Float16*)(ws + off); off = align256(off + (size_t)N_HID * N_GENES * 2);

    const int KT1 = N_TFS_P / 32;   // 47
    const int KTh = N_HID / 32;     // 2

    // pack weights -> f16 B-operand layout
    {
        size_t tot = (size_t)(N_HID / 16) * KT1 * 512;
        pack_b_kernel<<<(unsigned)((tot + 255) / 256), 256, 0, stream>>>(W1, W1p, N_TFS, N_HID, KT1);
    }
    {
        size_t tot = (size_t)(N_HID / 16) * KTh * 512;
        pack_b_kernel<<<(unsigned)((tot + 255) / 256), 256, 0, stream>>>(W2, W2p, N_HID, N_HID, KTh);
        pack_b_kernel<<<(unsigned)((tot + 255) / 256), 256, 0, stream>>>(Wd1, Wd1p, N_HID, N_HID, KTh);
    }
    {
        size_t tot = (size_t)(N_GENES / 16) * KTh * 512;
        pack_b_kernel<<<(unsigned)((tot + 255) / 256), 256, 0, stream>>>(Wd2, Wd2p, N_HID, N_GENES, KTh);
    }

    // sparse scatter + prelu(a0)
    scatter_kernel<<<BATCH / ROWS, 256, 0, stream>>>(x, gene_idx, tf_idx, edge_w, a0, h0);

    // encoder GEMM1
    gemm1_kernel<<<BATCH / 16, 128, 0, stream>>>(h0, W1p, b1, a1, h1);

    // encoder GEMM2 + emb + decoder GEMM1
    mid_kernel<<<BATCH / 16, 128, 0, stream>>>(h1, W2p, b2, a2, emb, pert, Wd1p, bd1, ad, h3);

    // final big GEMM
    dim3 g2(BATCH / 16, (N_GENES / 16 + 7) / 8);
    gemm2_kernel<<<g2, 256, 0, stream>>>(h3, Wd2p, bd2, out);
}
